// GINPlusLayer_67345087201309
// MI455X (gfx1250) — matrix operand order repaired and checked by reading, not compiled, not run on hardware
//
#include <hip/hip_runtime.h>

// ---------------------------------------------------------------------------
// GIN-plus attention layer for MI455X (gfx1250, wave32, WMMA f32_16x16x32_f16)
// K-outer GEMM loops: one LDS A-fragment feeds all of a wave's column tiles;
// B fragments for the whole burst are preloaded so s_wait_loadcnt staggers.
// ---------------------------------------------------------------------------

#define DEVFN static __device__ __forceinline__

typedef __attribute__((ext_vector_type(16))) _Float16 v16h;
typedef __attribute__((ext_vector_type(8)))  float    v8f;

static __device__ constexpr float kBnInv = 0.9999950000374997f; // 1/sqrt(1+1e-5)

union HalfPack { _Float16 h[2]; unsigned u; };
union AFragU   { v16h v; unsigned u[8]; };
union BFragU   { v16h v; uint4 q[2]; };

// A fragment (16x32 f16, M x K) from an LDS row-major tile, per ISA 7.12.2:
// lanes 0-15: M=lane, K pairs {2j,2j+1} (j<4) then {16+2j'}; lanes 16-31: +8 K.
DEVFN v16h lds_load_afrag(const _Float16* tile, int lda) {
  int lane = threadIdx.x & 31;
  int r    = lane & 15;
  int hi   = lane >> 4;
  const unsigned* rowp = reinterpret_cast<const unsigned*>(tile + r * lda);
  AFragU f;
#pragma unroll
  for (int j = 0; j < 8; ++j) {
    int k0 = (j < 4) ? (8 * hi + 2 * j) : (16 + 8 * hi + 2 * (j - 4));
    f.u[j] = rowp[k0 >> 1];
  }
  return f.v;
}

// B fragment (32x16 f16, K x N) from pre-packed weights: tile = 32 lanes x 8 dwords.
DEVFN v16h load_bfrag(const unsigned* wf, int tile_idx) {
  int lane = threadIdx.x & 31;
  const unsigned* p = wf + (size_t)tile_idx * 256 + (size_t)lane * 8;
  BFragU f;
  f.q[0] = *reinterpret_cast<const uint4*>(p);
  f.q[1] = *reinterpret_cast<const uint4*>(p + 4);
  return f.v;
}

DEVFN v8f wmma16(v16h a, v16h b, v8f c) {
  return __builtin_amdgcn_wmma_f32_16x16x32_f16(false, a, false, b, (short)0, c,
                                                false, false);
}

// ---------------------------------------------------------------------------
// Weight pre-pack: W[K,Nc] f32 (row-major) -> B-fragment tiles (f16 pairs).
// tile index = nt * (K/32) + kt ; lane n = (lane&15)+16*nt ; K = 32*kt+8*hi+2j (j<4), 32*kt+16+8*hi+2(j-4) (j>=4).
// ---------------------------------------------------------------------------
__global__ void pack_w_kernel(const float* __restrict__ W, unsigned* __restrict__ out,
                              int K, int Nc) {
  int tile = blockIdx.x;
  int nkt  = K >> 5;
  int kt   = tile % nkt;
  int nt   = tile / nkt;
  int lane = threadIdx.x;           // 32 threads
  int n    = (lane & 15) + (nt << 4);
  int hi   = lane >> 4;
  unsigned* dst = out + (size_t)tile * 256 + (size_t)lane * 8;
#pragma unroll
  for (int j = 0; j < 8; ++j) {
    int k0 = (kt << 5) + ((j < 4) ? (8 * hi + 2 * j) : (16 + 8 * hi + 2 * (j - 4)));
    HalfPack p;
    p.h[0] = (_Float16)W[(size_t)k0 * Nc + n];
    p.h[1] = (_Float16)W[(size_t)(k0 + 1) * Nc + n];
    dst[j] = p.u;
  }
}

// ---------------------------------------------------------------------------
// init: zero agg + softmax scalars
// ---------------------------------------------------------------------------
__global__ void init_kernel(float* __restrict__ agg, unsigned* __restrict__ maxkey,
                            float* __restrict__ gsum, size_t n) {
  size_t i = (size_t)blockIdx.x * blockDim.x + threadIdx.x;
  if (i == 0) { *maxkey = 0u; *gsum = 0.0f; }
  for (; i < n; i += (size_t)gridDim.x * blockDim.x) agg[i] = 0.0f;
}

// ---------------------------------------------------------------------------
// Fused encoder: O = relu(bn(A[M,K] @ W[K,256] + b)).  128 thr = 4 waves,
// each wave 4 column tiles; K-outer, B-burst preloaded.
// ---------------------------------------------------------------------------
template <int K>
__global__ void encoder_kernel(const float* __restrict__ A,
                               const unsigned* __restrict__ wf,
                               const float* __restrict__ bias,
                               const float* __restrict__ g,
                               const float* __restrict__ beta,
                               float* __restrict__ out_f32,
                               _Float16* __restrict__ out_f16, int M) {
  __shared__ _Float16 at[16 * K];
  const int r0  = blockIdx.x << 4;
  const int tid = threadIdx.x;
  if (r0 >= M) return;

  for (int idx = tid; idx < 16 * K; idx += 128) {
    int r = idx / K, c = idx % K;
    at[idx] = (_Float16)A[(size_t)(r0 + r) * K + c];
  }
  __syncthreads();

  const int wave = tid >> 5, lane = tid & 31;
  const int n = lane & 15, hi = lane >> 4;
  const int NKT = K / 32;

  v8f acc[4] = {};
#pragma unroll
  for (int kt = 0; kt < NKT; ++kt) {
    v16h a = lds_load_afrag(at + kt * 32, K);
    __builtin_prefetch(wf + ((size_t)(wave * 4 * NKT + kt + 1) * 256), 0, 1);
    v16h b[4];
#pragma unroll
    for (int t = 0; t < 4; ++t)
      b[t] = load_bfrag(wf, (wave * 4 + t) * NKT + kt);
#pragma unroll
    for (int t = 0; t < 4; ++t)
      acc[t] = wmma16(a, b[t], acc[t]);
  }

#pragma unroll
  for (int t = 0; t < 4; ++t) {
    int c = (wave * 4 + t) * 16 + n;
    float s = g[c] * kBnInv, bb = beta[c], bi = bias[c];
#pragma unroll
    for (int i = 0; i < 8; ++i) {
      int m = i + 8 * hi;
      float v = acc[t][i] + bi;
      v = v * s + bb;
      v = v > 0.0f ? v : 0.0f;
      size_t o = (size_t)(r0 + m) * 256 + c;
      if (out_f32) out_f32[o] = v;
      if (out_f16) out_f16[o] = (_Float16)v;
    }
  }
}

// ---------------------------------------------------------------------------
// Attention logits: per 16-edge tile build att_in = [h[row]+ef | h[col]] (f16,
// 16x512 in LDS), WMMA vs Wa1 (K-outer, B-burst preloaded), tanh, dot Wa2 via
// LDS float atomics, leaky-relu, softmax-max via monotone-key atomicMax.
// ---------------------------------------------------------------------------
__global__ void attn_kernel(const long long* __restrict__ ei,
                            const float* __restrict__ h,
                            const _Float16* __restrict__ ef,
                            const unsigned* __restrict__ wa1f,
                            const float* __restrict__ ba1,
                            const float* __restrict__ wa2,
                            const float* __restrict__ ba2,
                            float* __restrict__ logits,
                            unsigned* __restrict__ gmaxkey, int E) {
  __shared__ _Float16 att[16][512];
  __shared__ float scores[16];
  __shared__ int rows16[16], cols16[16];
  const int e0  = blockIdx.x << 4;
  const int tid = threadIdx.x;

  if (tid < 16) {
    rows16[tid] = (int)ei[e0 + tid];
    cols16[tid] = (int)ei[(size_t)E + e0 + tid];
    scores[tid] = ba2[0];
  }
  __syncthreads();

  for (int idx = tid; idx < 16 * 512; idx += 128) {
    int r = idx >> 9, c = idx & 511;
    float v;
    if (c < 256)
      v = h[(size_t)rows16[r] * 256 + c] + (float)ef[(size_t)(e0 + r) * 256 + c];
    else
      v = h[(size_t)cols16[r] * 256 + (c - 256)];
    att[r][c] = (_Float16)v;
  }
  __syncthreads();

  const int wave = tid >> 5, lane = tid & 31;
  const int n = lane & 15, hi = lane >> 4;

  v8f acc[4] = {};
#pragma unroll
  for (int kt = 0; kt < 16; ++kt) {
    v16h a = lds_load_afrag(&att[0][0] + kt * 32, 512);
    __builtin_prefetch(wa1f + ((size_t)(wave * 64 + kt + 1) * 256), 0, 1);
    v16h b[4];
#pragma unroll
    for (int t = 0; t < 4; ++t)
      b[t] = load_bfrag(wa1f, (wave * 4 + t) * 16 + kt);
#pragma unroll
    for (int t = 0; t < 4; ++t)
      acc[t] = wmma16(a, b[t], acc[t]);
  }

#pragma unroll
  for (int t = 0; t < 4; ++t) {
    int c = (wave * 4 + t) * 16 + n;
    float bi = ba1[c], w2 = wa2[c];
#pragma unroll
    for (int i = 0; i < 8; ++i) {
      int m = i + 8 * hi;
      float tv = tanhf(acc[t][i] + bi);
      atomicAdd(&scores[m], tv * w2);   // ds_add_f32
    }
  }
  __syncthreads();

  if (tid < 16) {
    float a = scores[tid];
    a = a > 0.0f ? a : 0.2f * a;        // leaky_relu(0.2)
    logits[e0 + tid] = a;
    unsigned bits = __float_as_uint(a);
    unsigned key  = (bits & 0x80000000u) ? ~bits : (bits | 0x80000000u);
    atomicMax(gmaxkey, key);            // monotone-ordered float max
  }
}

// ---------------------------------------------------------------------------
// Softmax denominator: sum(exp(l - max)) -> *gsum
// ---------------------------------------------------------------------------
__global__ void softmax_sum_kernel(const float* __restrict__ logits,
                                   const unsigned* __restrict__ gmaxkey,
                                   float* __restrict__ gsum, int E) {
  __shared__ float red[8];
  unsigned key  = *gmaxkey;
  unsigned bits = (key & 0x80000000u) ? (key & 0x7FFFFFFFu) : ~key;
  float mx = __uint_as_float(bits);
  float s  = 0.0f;
  for (int i = blockIdx.x * blockDim.x + threadIdx.x; i < E;
       i += gridDim.x * blockDim.x)
    s += __expf(logits[i] - mx);
#pragma unroll
  for (int off = 16; off > 0; off >>= 1) s += __shfl_down(s, off, 32);
  if ((threadIdx.x & 31) == 0) red[threadIdx.x >> 5] = s;
  __syncthreads();
  if (threadIdx.x < 8) {
    float v = red[threadIdx.x];
#pragma unroll
    for (int off = 4; off > 0; off >>= 1) v += __shfl_down(v, off, 8);
    if (threadIdx.x == 0) atomicAdd(gsum, v);
  }
}

// ---------------------------------------------------------------------------
// Messages: msg = (h[col] + ef) * softmax(a); scatter-add into agg[row].
// 64 threads per edge, 4 columns each.
// ---------------------------------------------------------------------------
__global__ void message_kernel(const long long* __restrict__ ei,
                               const float* __restrict__ h,
                               const _Float16* __restrict__ ef,
                               const float* __restrict__ logits,
                               const unsigned* __restrict__ gmaxkey,
                               const float* __restrict__ gsum,
                               float* __restrict__ agg, int E) {
  long long gid = (long long)blockIdx.x * blockDim.x + threadIdx.x;
  int e = (int)(gid >> 6);
  if (e >= E) return;
  int c0 = (int)(gid & 63) << 2;
  unsigned key  = *gmaxkey;
  unsigned bits = (key & 0x80000000u) ? (key & 0x7FFFFFFFu) : ~key;
  float a = __expf(logits[e] - __uint_as_float(bits)) / *gsum;
  size_t r = (size_t)ei[e];
  size_t c = (size_t)ei[(size_t)E + e];
#pragma unroll
  for (int k = 0; k < 4; ++k) {
    int cc = c0 + k;
    float m = (h[c * 256 + cc] + (float)ef[(size_t)e * 256 + cc]) * a;
    atomicAdd(&agg[r * 256 + cc], m);   // global_atomic_add_f32
  }
}

// ---------------------------------------------------------------------------
// Final fused MLP head: out=(1+eps)h+agg; mid=relu(bn(out@W1+b1)); y=mid@W2+b2
// Two chained WMMA stages through LDS, both K-outer with B-burst preload.
// ---------------------------------------------------------------------------
__global__ void final_mlp_kernel(const float* __restrict__ h,
                                 const float* __restrict__ agg,
                                 const float* __restrict__ eps,
                                 const unsigned* __restrict__ w1f,
                                 const float* __restrict__ b1,
                                 const float* __restrict__ gm,
                                 const float* __restrict__ betam,
                                 const unsigned* __restrict__ w2f,
                                 const float* __restrict__ b2,
                                 float* __restrict__ out, int N) {
  __shared__ _Float16 a0[16][256];
  __shared__ _Float16 a1[16][512];
  const int r0  = blockIdx.x << 4;
  const int tid = threadIdx.x;
  if (r0 >= N) return;
  const float ep = 1.0f + eps[0];

  for (int idx = tid; idx < 16 * 256; idx += 128) {
    int r = idx >> 8, c = idx & 255;
    size_t o = (size_t)(r0 + r) * 256 + c;
    a0[r][c] = (_Float16)(ep * h[o] + agg[o]);
  }
  __syncthreads();

  const int wave = tid >> 5, lane = tid & 31;
  const int n = lane & 15, hi = lane >> 4;

  // stage 1: mid[16,512] = relu(bn(a0 @ W1 + b1)), K=256 (8 k-tiles), 8 col
  // tiles per wave; pairwise B preload to bound register pressure.
  {
    v8f acc[8] = {};
#pragma unroll
    for (int kt = 0; kt < 8; ++kt) {
      v16h a = lds_load_afrag(&a0[0][0] + kt * 32, 256);
#pragma unroll
      for (int tp = 0; tp < 4; ++tp) {
        v16h b0 = load_bfrag(w1f, (wave * 8 + 2 * tp) * 8 + kt);
        v16h b1v = load_bfrag(w1f, (wave * 8 + 2 * tp + 1) * 8 + kt);
        acc[2 * tp]     = wmma16(a, b0,  acc[2 * tp]);
        acc[2 * tp + 1] = wmma16(a, b1v, acc[2 * tp + 1]);
      }
    }
#pragma unroll
    for (int t = 0; t < 8; ++t) {
      int c = (wave * 8 + t) * 16 + n;
      float s = gm[c] * kBnInv, bb = betam[c], bi = b1[c];
#pragma unroll
      for (int i = 0; i < 8; ++i) {
        int m = i + 8 * hi;
        float v = acc[t][i] + bi;
        v = v * s + bb;
        v = v > 0.0f ? v : 0.0f;
        a1[m][c] = (_Float16)v;
      }
    }
  }
  __syncthreads();

  // stage 2: y[16,256] = mid @ W2 + b2, K=512 (16 k-tiles), 4 col tiles/wave.
  {
    v8f acc[4] = {};
#pragma unroll
    for (int kt = 0; kt < 16; ++kt) {
      v16h a = lds_load_afrag(&a1[0][0] + kt * 32, 512);
      v16h b[4];
#pragma unroll
      for (int t = 0; t < 4; ++t)
        b[t] = load_bfrag(w2f, (wave * 4 + t) * 16 + kt);
#pragma unroll
      for (int t = 0; t < 4; ++t)
        acc[t] = wmma16(a, b[t], acc[t]);
    }
#pragma unroll
    for (int t = 0; t < 4; ++t) {
      int c = (wave * 4 + t) * 16 + n;
      float bi = b2[c];
#pragma unroll
      for (int i = 0; i < 8; ++i)
        out[(size_t)(r0 + i + 8 * hi) * 256 + c] = acc[t][i] + bi;
    }
  }
}

// ---------------------------------------------------------------------------
// Host launcher
// ---------------------------------------------------------------------------
extern "C" void kernel_launch(void* const* d_in, const int* in_sizes, int n_in,
                              void* d_out, int out_size, void* d_ws, size_t ws_size,
                              hipStream_t stream) {
  (void)in_sizes; (void)n_in; (void)out_size; (void)ws_size;
  const int N = 50000, E = 400000;

  const float*     x    = (const float*)d_in[0];
  const long long* ei   = (const long long*)d_in[1];   // int64 edge_index [2,E]
  const float*     ea   = (const float*)d_in[2];
  const float*     Wn   = (const float*)d_in[3];
  const float*     bn_b = (const float*)d_in[4];
  const float*     gn   = (const float*)d_in[5];
  const float*     betan= (const float*)d_in[6];
  const float*     We   = (const float*)d_in[7];
  const float*     be   = (const float*)d_in[8];
  const float*     ge   = (const float*)d_in[9];
  const float*     betae= (const float*)d_in[10];
  const float*     Wa1  = (const float*)d_in[11];
  const float*     ba1  = (const float*)d_in[12];
  const float*     Wa2  = (const float*)d_in[13];
  const float*     ba2  = (const float*)d_in[14];
  const float*     W1   = (const float*)d_in[15];
  const float*     b1   = (const float*)d_in[16];
  const float*     gm   = (const float*)d_in[17];
  const float*     betam= (const float*)d_in[18];
  const float*     W2   = (const float*)d_in[19];
  const float*     b2   = (const float*)d_in[20];
  const float*     eps  = (const float*)d_in[21];
  float*           y    = (float*)d_out;

  // workspace carve-out
  char* ws = (char*)d_ws;
  size_t off = 0;
  auto carve = [&](size_t bytes) -> void* {
    void* p = ws + off;
    off += (bytes + 255) & ~(size_t)255;
    return p;
  };
  float*     h      = (float*)carve((size_t)N * 256 * 4);
  float*     agg    = (float*)carve((size_t)N * 256 * 4);
  _Float16*  ef     = (_Float16*)carve((size_t)E * 256 * 2);
  float*     logits = (float*)carve((size_t)E * 4);
  unsigned*  wnf    = (unsigned*)carve((size_t)64  * 1024);
  unsigned*  wef    = (unsigned*)carve((size_t)16  * 1024);
  unsigned*  wa1f   = (unsigned*)carve((size_t)256 * 1024);
  unsigned*  w1f    = (unsigned*)carve((size_t)256 * 1024);
  unsigned*  w2f    = (unsigned*)carve((size_t)256 * 1024);
  unsigned*  maxkey = (unsigned*)carve(256);
  float*     gsum   = (float*)carve(256);

  // 0) zero agg + softmax scalars
  init_kernel<<<2048, 256, 0, stream>>>(agg, maxkey, gsum, (size_t)N * 256);

  // 1) pre-pack all weights into WMMA B-fragment layout
  pack_w_kernel<<<64,  32, 0, stream>>>(Wn,  wnf,  128, 256);
  pack_w_kernel<<<16,  32, 0, stream>>>(We,  wef,   32, 256);
  pack_w_kernel<<<256, 32, 0, stream>>>(Wa1, wa1f, 512, 256);
  pack_w_kernel<<<256, 32, 0, stream>>>(W1,  w1f,  256, 512);
  pack_w_kernel<<<256, 32, 0, stream>>>(W2,  w2f,  512, 256);

  // 2) encoders (node -> h f32; edge -> ef f16)
  encoder_kernel<128><<<N / 16, 128, 0, stream>>>(x,  wnf, bn_b, gn, betan, h,  nullptr, N);
  encoder_kernel<32> <<<E / 16, 128, 0, stream>>>(ea, wef, be,   ge, betae, nullptr, ef, E);

  // 3) attention logits + global max
  attn_kernel<<<E / 16, 128, 0, stream>>>(ei, h, ef, wa1f, ba1, Wa2, ba2,
                                          logits, maxkey, E);

  // 4) softmax denominator
  softmax_sum_kernel<<<512, 256, 0, stream>>>(logits, maxkey, gsum, E);

  // 5) weighted messages + scatter-add
  message_kernel<<<(unsigned)(((size_t)E * 64 + 255) / 256), 256, 0, stream>>>(
      ei, h, ef, logits, maxkey, gsum, agg, E);

  // 6) fused MLP head -> d_out
  final_mlp_kernel<<<N / 16, 128, 0, stream>>>(h, agg, eps, w1f, b1, gm, betam,
                                               w2f, b2, y, N);
}
